// MidBlock_30915174596716
// MI455X (gfx1250) — matrix operand order repaired and checked by reading, not compiled, not run on hardware
//
#include <hip/hip_runtime.h>
#include <math.h>

typedef __attribute__((ext_vector_type(16))) _Float16 v16h;
typedef __attribute__((ext_vector_type(8)))  _Float16 v8h;
typedef __attribute__((ext_vector_type(8)))  float    v8f;

// ---------------- activation helpers ----------------
__device__ __forceinline__ float siluf(float v) { return v / (1.f + __expf(-v)); }
__device__ __forceinline__ float geluf(float v) {
    const float c = 0.7978845608028654f;
    float t = tanhf(c * (v + 0.044715f * v * v * v));
    return 0.5f * v * (1.f + t);
}
__device__ __forceinline__ float softplusf(float v) {
    return v > 0.f ? v + log1pf(__expf(-v)) : log1pf(__expf(v));
}
__device__ __forceinline__ float applyAct(float v, int act) {
    if (act == 1) return geluf(v);
    if (act == 2) return softplusf(v);
    return v;
}
__device__ __forceinline__ int imin(int a, int b) { return a < b ? a : b; }
__device__ __forceinline__ int imax(int a, int b) { return a > b ? a : b; }

// ---------------- WMMA fragment loaders (contiguous LDS -> 2x ds_load_b128) ------
__device__ __forceinline__ v16h cat16(v8h lo, v8h hi) {
    return __builtin_shufflevector(lo, hi, 0, 1, 2, 3, 4, 5, 6, 7, 8, 9, 10, 11, 12, 13,
                                   14, 15);
}
// sA row-major [32][32] halves. A 16x32 frag: m=lane&15(+16*mhalf), e<8 -> k=8*hi+e,
// e>=8 -> k=16+8*hi+(e-8)  => two contiguous 8-half chunks.
__device__ __forceinline__ v16h fragA(const _Float16* sA, int lane, int mhalf) {
    int m = (lane & 15) + mhalf * 16;
    int hi = (lane >> 4) & 1;
    const v8h* p = (const v8h*)(sA + m * 32 + 8 * hi);
    return cat16(p[0], p[2]);
}
// sBt transposed [n][k] [32][32]. B 32x16 frag: n=lane&15(+16*nhalf), e<8 -> k=8*hi+e,
// e>=8 -> k=16+8*hi+(e-8)  => two contiguous 8-half chunks.
__device__ __forceinline__ v16h fragB(const _Float16* sBt, int lane, int nhalf) {
    int n = (lane & 15) + nhalf * 16;
    int hi = (lane >> 4) & 1;
    const v8h* p = (const v8h*)(sBt + n * 32 + 8 * hi);
    return cat16(p[0], p[2]);
}

// Fast path: 32 contiguous K-values, 16B-aligned -> 8x global_load_b128.
__device__ __forceinline__ void stage32_vec(const float* __restrict__ base, int off,
                                            int k0, _Float16* dst) {
    const float4* p = (const float4*)(base + off + k0);
#pragma unroll
    for (int c = 0; c < 4; ++c) {
        float4 u = p[2 * c];
        float4 v = p[2 * c + 1];
        v8h pk;
        pk[0] = (_Float16)u.x; pk[1] = (_Float16)u.y;
        pk[2] = (_Float16)u.z; pk[3] = (_Float16)u.w;
        pk[4] = (_Float16)v.x; pk[5] = (_Float16)v.y;
        pk[6] = (_Float16)v.z; pk[7] = (_Float16)v.w;
        ((v8h*)dst)[c] = pk;
    }
}

// Strided / guarded path: independent 32-bit indices per element (no addr chain).
template <bool GUARD>
__device__ __forceinline__ void stage32s(const float* __restrict__ base, int off,
                                         int stride, int k0, int Kd, _Float16* dst) {
#pragma unroll
    for (int c = 0; c < 4; ++c) {
        unsigned idx[8];
#pragma unroll
        for (int e = 0; e < 8; ++e) {
            int kg = k0 + c * 8 + e;
            int kc = GUARD ? imin(kg, Kd - 1) : kg;
            idx[e] = (unsigned)(off + kc * stride);
        }
        float t[8];
#pragma unroll
        for (int e = 0; e < 8; ++e) t[e] = base[idx[e]];
        v8h pk;
#pragma unroll
        for (int e = 0; e < 8; ++e) {
            int kg = k0 + c * 8 + e;
            float v = (!GUARD || kg < Kd) ? t[e] : 0.f;
            pk[e] = (_Float16)v;
        }
        ((v8h*)dst)[c] = pk;
    }
}

__device__ __forceinline__ void stageAny(const float* __restrict__ base, int off,
                                         int stride, int k0, int Kd, _Float16* dst) {
    if (stride == 1 && k0 + 32 <= Kd) {
        stage32_vec(base, off, k0, dst);
    } else if (k0 + 32 <= Kd) {
        stage32s<false>(base, off, stride, k0, Kd, dst);
    } else {
        stage32s<true>(base, off, stride, k0, Kd, dst);
    }
}

#define WMMA4(acc, A0, A1, B0, B1)                                                        \
    acc[0][0] = __builtin_amdgcn_wmma_f32_16x16x32_f16(false, A0, false, B0, (short)0,    \
                                                       acc[0][0], false, false);          \
    acc[0][1] = __builtin_amdgcn_wmma_f32_16x16x32_f16(false, A0, false, B1, (short)0,    \
                                                       acc[0][1], false, false);          \
    acc[1][0] = __builtin_amdgcn_wmma_f32_16x16x32_f16(false, A1, false, B0, (short)0,    \
                                                       acc[1][0], false, false);          \
    acc[1][1] = __builtin_amdgcn_wmma_f32_16x16x32_f16(false, A1, false, B1, (short)0,    \
                                                       acc[1][1], false, false)

// ---------------- generic strided-batched GEMM: C = act(A*B + bias) + resid ------
// One wave computes a 32x32 C tile (4 WMMAs/chunk). grid=(M/32, ceil(N/32), batch)
__global__ __launch_bounds__(32) void k_gemm_wmma(
    const float* __restrict__ A, const float* __restrict__ B, float* __restrict__ C,
    const float* __restrict__ bias, const float* __restrict__ resid, int Kd, int N,
    int aRow, int aCol, int aBatch, int bRow, int bCol, int bBatch, int bMod, int cRow,
    int cCol, int cBatch, int biasBatch, int act) {
    __shared__ _Float16 sA[32 * 32];
    __shared__ _Float16 sBt[32 * 32];
    int lane = threadIdx.x;
    int row0 = blockIdx.x * 32, col0 = blockIdx.y * 32, bz = blockIdx.z;
    const float* A2 = A + (long)bz * aBatch;          // uniform base
    int aOff = (row0 + lane) * aRow;                  // 32-bit lane index
    const float* B2 = B + (long)(bz % bMod) * bBatch;
    int bOff = imin(col0 + lane, N - 1) * bCol;
    float* Cb = C + (long)bz * cBatch;
    const float* Rb = resid ? resid + (long)bz * cBatch : nullptr;
    const float* biasB = bias ? bias + (long)(bz % bMod) * biasBatch : nullptr;
    v8f acc[2][2] = {};
    for (int k0 = 0; k0 < Kd; k0 += 32) {
        stageAny(A2, aOff, aCol, k0, Kd, sA + lane * 32);
        stageAny(B2, bOff, bRow, k0, Kd, sBt + lane * 32);
        __syncthreads();
        v16h A0 = fragA(sA, lane, 0), A1 = fragA(sA, lane, 1);
        v16h B0 = fragB(sBt, lane, 0), B1 = fragB(sBt, lane, 1);
        WMMA4(acc, A0, A1, B0, B1);
        __syncthreads();
    }
    int nl = lane & 15, hi = (lane >> 4) & 1;
#pragma unroll
    for (int mi = 0; mi < 2; ++mi)
#pragma unroll
        for (int ni = 0; ni < 2; ++ni) {
            int col = col0 + ni * 16 + nl;
            if (col < N) {
#pragma unroll
                for (int r = 0; r < 8; ++r) {
                    int m = row0 + mi * 16 + r + 8 * hi;
                    int ro = m * cRow + col * cCol;   // < 2^31 for all our shapes
                    float v = acc[mi][ni][r];
                    if (biasB) v += biasB[col];
                    v = applyAct(v, act);
                    if (Rb) v += Rb[ro];
                    Cb[ro] = v;
                }
            }
        }
}

// ---------------- 3x3 conv 256->256 implicit GEMM (K=2304), 32x32 tile -----------
__global__ __launch_bounds__(32) void k_conv3x3_wmma(
    const float* __restrict__ in, const float* __restrict__ w,
    const float* __restrict__ bias, const float* __restrict__ resid,
    float* __restrict__ out) {
    __shared__ _Float16 sA[32 * 32];
    __shared__ _Float16 sBt[32 * 32];
    int lane = threadIdx.x;
    int row0 = blockIdx.x * 32, col0 = blockIdx.y * 32;
    int b = row0 >> 10;                    // uniform: tile never crosses batch
    int hw0 = row0 & 1023;                 // uniform base pixel within batch
    int hwl = hw0 + lane;
    int h = hwl >> 5, x = hwl & 31;
    const float* inb = in + ((long)b << 18);            // + b*256*1024
    const float* w2 = w + (long)col0 * 2304;            // w[n][kg], kg contiguous
    int wOff = lane * 2304;
    v8f acc[2][2] = {};
    for (int k0 = 0; k0 < 2304; k0 += 32) {
        // B: weights are K-contiguous and 16B aligned -> vector path
        stage32_vec(w2, wOff, k0, sBt + lane * 32);
        // A: 9-tap gather with branch-free bounds handling
#pragma unroll
        for (int c = 0; c < 4; ++c) {
            unsigned idx[8];
            bool okm[8];
#pragma unroll
            for (int e = 0; e < 8; ++e) {
                int kg = k0 + c * 8 + e;   // kg = ci*9 + tap
                int ci = kg / 9, tap = kg - ci * 9;
                int y = h + tap / 3 - 1, xx = x + (tap % 3) - 1;
                okm[e] = ((unsigned)y < 32u) & ((unsigned)xx < 32u);
                idx[e] = (unsigned)((ci << 10) + imin(imax(y * 32 + xx, 0), 1023));
            }
            float t[8];
#pragma unroll
            for (int e = 0; e < 8; ++e) t[e] = inb[idx[e]];
            v8h pa;
#pragma unroll
            for (int e = 0; e < 8; ++e) pa[e] = (_Float16)(okm[e] ? t[e] : 0.f);
            ((v8h*)(sA + lane * 32))[c] = pa;
        }
        __syncthreads();
        v16h A0 = fragA(sA, lane, 0), A1 = fragA(sA, lane, 1);
        v16h B0 = fragB(sBt, lane, 0), B1 = fragB(sBt, lane, 1);
        WMMA4(acc, A0, A1, B0, B1);
        __syncthreads();
    }
    int nl = lane & 15, hi = (lane >> 4) & 1;
#pragma unroll
    for (int mi = 0; mi < 2; ++mi)
#pragma unroll
        for (int ni = 0; ni < 2; ++ni) {
            int n = col0 + ni * 16 + nl;
#pragma unroll
            for (int r = 0; r < 8; ++r) {
                int phw = hw0 + mi * 16 + r + 8 * hi;
                int o = ((b * 256 + n) << 10) + phw;
                float v = acc[mi][ni][r] + bias[n];
                if (resid) v += resid[o];
                out[o] = v;
            }
        }
}

// ---------------- block reductions ----------------
template <int BS>
__device__ __forceinline__ void blockReduce2(float& a, float& b) {
    __shared__ float r1[BS];
    __shared__ float r2[BS];
    int t = threadIdx.x;
    r1[t] = a; r2[t] = b;
    __syncthreads();
    for (int off = BS / 2; off > 0; off >>= 1) {
        if (t < off) { r1[t] += r1[t + off]; r2[t] += r2[t + off]; }
        __syncthreads();
    }
    a = r1[0]; b = r2[0];
    __syncthreads();
}

// ---------------- GroupNorm(32) + SiLU over (B,256,32,32): block per (b,g) -------
__global__ void k_groupnorm_silu(const float* __restrict__ in, const float* __restrict__ s,
                                 const float* __restrict__ bb, float* __restrict__ out) {
    int b = blockIdx.x >> 5, g = blockIdx.x & 31;
    long base = ((long)(b * 256 + g * 8)) << 10;
    float sum = 0.f, sq = 0.f;
    for (int i = threadIdx.x; i < 8192; i += 256) {
        float v = in[base + i];
        sum += v; sq += v * v;
    }
    blockReduce2<256>(sum, sq);
    float mean = sum * (1.f / 8192.f);
    float var = sq * (1.f / 8192.f) - mean * mean;
    float rs = rsqrtf(var + 1e-6f);
    for (int i = threadIdx.x; i < 8192; i += 256) {
        int c = g * 8 + (i >> 10);
        float v = (in[base + i] - mean) * rs * s[c] + bb[c];
        out[base + i] = siluf(v);
    }
}

// ---------------- LayerNorm over last dim (+ optional SiLU gate) -----------------
__global__ void k_layernorm(const float* __restrict__ in, const float* __restrict__ s,
                            const float* __restrict__ bb, float* __restrict__ out, int dim,
                            int rowStride, const float* __restrict__ gate, int gStride,
                            int gOff) {
    long row = blockIdx.x;
    const float* xr = in + row * rowStride;
    float sum = 0.f, sq = 0.f;
    for (int i = threadIdx.x; i < dim; i += 128) {
        float v = xr[i];
        sum += v; sq += v * v;
    }
    blockReduce2<128>(sum, sq);
    float inv = 1.f / (float)dim;
    float mean = sum * inv;
    float var = sq * inv - mean * mean;
    float rs = rsqrtf(var + 1e-5f);
    for (int i = threadIdx.x; i < dim; i += 128) {
        float v = (xr[i] - mean) * rs * s[i] + bb[i];
        if (gate) {
            float z = gate[row * (long)gStride + gOff + i];
            v *= siluf(z);
        }
        out[row * (long)dim + i] = v;
    }
}

// ---------------- depthwise 3x3 conv + SiLU; out xconv (B,L,512) channel-fastest --
__global__ void k_dwconv_silu(const float* __restrict__ xz, const float* __restrict__ w,
                              const float* __restrict__ bias, float* __restrict__ out) {
    int t = blockIdx.x * blockDim.x + threadIdx.x;   // 8*1024*512, d fastest
    int b = t >> 19, rem = t & ((1 << 19) - 1);
    int hw = rem >> 9, d = rem & 511;
    int h = hw >> 5, x = hw & 31;
    float acc = bias[d];
#pragma unroll
    for (int dy = -1; dy <= 1; ++dy)
#pragma unroll
        for (int dx = -1; dx <= 1; ++dx) {
            int y = h + dy, xx = x + dx;
            if ((unsigned)y < 32u && (unsigned)xx < 32u)
                acc += xz[(((long)b << 10) + y * 32 + xx) * 1024 + d] *
                       w[d * 9 + (dy + 1) * 3 + (dx + 1)];
        }
    out[(((long)b << 10) + hw) * 512 + d] = siluf(acc);
}

// ---------------- build xs (B,K,L,512) from xconv (B,L,512) ----------------------
__global__ void k_build_xs(const float* __restrict__ xc, float* __restrict__ xs) {
    int t = blockIdx.x * blockDim.x + threadIdx.x;   // 8*1024*512, d fastest
    int b = t >> 19, rem = t & ((1 << 19) - 1);
    int l = rem >> 9, d = rem & 511;
    float v = xc[(((long)b << 10) + l) * 512 + d];
    int h = l >> 5, w = l & 31, l1 = (w << 5) | h;
    long base = ((long)(b * 4)) << 19;               // *1024*512
    const long s = 1L << 19;
    xs[base + (long)l * 512 + d] = v;
    xs[base + s + (long)l1 * 512 + d] = v;
    xs[base + 2 * s + (long)(1023 - l) * 512 + d] = v;
    xs[base + 3 * s + (long)(1023 - l1) * 512 + d] = v;
}

// ---------------- selective scan: thread per (b,k,d); lanes coalesced in d -------
// xs,dy layout (B,K,L,512); ys overwrites delta (dy) in place
__global__ void k_scan(const float* __restrict__ xs, float* __restrict__ dy,
                       const float* __restrict__ xdbl, const float* __restrict__ A_logs,
                       const float* __restrict__ Ds) {
    int t = blockIdx.x * blockDim.x + threadIdx.x;   // 16384
    int b = t >> 11, rem = t & 2047;
    int k = rem >> 9, d = rem & 511;
    long base = ((long)(b * 4 + k)) << 19;
    const float* xsb = xs + base + d;
    float* dyb = dy + base + d;
    long g2 = (long)(b * 4 + k) * 48;
    const float* Bsb = xdbl + ((g2 + 16) << 10);
    const float* Csb = xdbl + ((g2 + 32) << 10);
    const float* Arow = A_logs + (long)(k * 512 + d) * 16;
    float Av[16], hs[16];
#pragma unroll
    for (int n = 0; n < 16; ++n) { Av[n] = -__expf(Arow[n]); hs[n] = 0.f; }
    float Dv = Ds[k * 512 + d];
    for (int l = 0; l < 1024; ++l) {
        float dlt = dyb[l * 512];
        float u = xsb[l * 512];
        float du = dlt * u;
        float y = 0.f;
#pragma unroll
        for (int n = 0; n < 16; ++n) {
            hs[n] = __expf(dlt * Av[n]) * hs[n] + du * Bsb[(n << 10) + l];
            y += hs[n] * Csb[(n << 10) + l];
        }
        dyb[l * 512] = y + u * Dv;
    }
}

// ---------------- merge 4 directions (B,K,L,512) -> (B,L,512) --------------------
__global__ void k_merge(const float* __restrict__ ys, float* __restrict__ yb) {
    int t = blockIdx.x * blockDim.x + threadIdx.x;   // 8*1024*512, d fastest
    int b = t >> 19, rem = t & ((1 << 19) - 1);
    int l = rem >> 9, d = rem & 511;
    int h = l >> 5, w = l & 31, l1 = (w << 5) | h;
    long base = ((long)(b * 4)) << 19;
    const long s = 1L << 19;
    float y = ys[base + (long)l * 512 + d] + ys[base + 2 * s + (long)(1023 - l) * 512 + d] +
              ys[base + s + (long)l1 * 512 + d] +
              ys[base + 3 * s + (long)(1023 - l1) * 512 + d];
    yb[(((long)b << 10) + l) * 512 + d] = y;
}

// ---------------- layout transposes ----------------------------------------------
__global__ void k_bchw_to_bhwc(const float* __restrict__ in, float* __restrict__ out) {
    int t = blockIdx.x * blockDim.x + threadIdx.x;   // 2097152
    int b = t >> 18, rem = t & 262143;
    int c = rem >> 10, hw = rem & 1023;
    out[(((long)b << 10) + hw) * 256 + c] = in[t];
}
__global__ void k_bhwc_to_bchw(const float* __restrict__ in, float* __restrict__ out) {
    int t = blockIdx.x * blockDim.x + threadIdx.x;   // 2097152
    int b = t >> 18, rem = t & 262143;
    int hw = rem >> 8, c = rem & 255;
    out[(((long)(b * 256 + c)) << 10) + hw] = in[t];
}

// =================================================================================
extern "C" void kernel_launch(void* const* d_in, const int* in_sizes, int n_in,
                              void* d_out, int out_size, void* d_ws, size_t ws_size,
                              hipStream_t stream) {
    (void)in_sizes; (void)n_in; (void)out_size; (void)ws_size;
    auto in = [&](int i) { return (const float*)d_in[i]; };
    char* ws = (char*)d_ws;
    const size_t MB = 1u << 20;
    float* W0 = (float*)(ws + 0 * MB);     // 8MB  BCHW ping
    float* W1 = (float*)(ws + 8 * MB);     // 8MB  BCHW pong
    float* W2 = (float*)(ws + 16 * MB);    // 8MB  gn/silu tmp
    float* W3 = (float*)(ws + 24 * MB);    // 8MB  conv1 tmp
    float* W4 = (float*)(ws + 32 * MB);    // 8MB  BHWC x (vss residual stream)
    float* W5 = (float*)(ws + 40 * MB);    // 8MB  ln tmp (8192x256)
    float* W6 = (float*)(ws + 48 * MB);    // 32MB xz / MLP hidden
    float* W7 = (float*)(ws + 80 * MB);    // 64MB xs (B,K,L,512)
    float* W8 = (float*)(ws + 144 * MB);   // 64MB delta -> ys (B,K,L,512), in-place
    float* W9 = (float*)(ws + 208 * MB);   // 8MB  x_dbl (B,K,48,L)
    float* W10 = (float*)(ws + 216 * MB);  // 16MB xconv (B,L,512), then ybuf (B,L,512)
    float* W11 = (float*)(ws + 232 * MB);  // 16MB gated out-norm (B,L,512)

    auto gemm = [&](const float* A, const float* B, float* C, const float* bias,
                    const float* resid, int M, int N, int K, int aR, int aC, int aB,
                    int bR, int bC, int bB, int bMod, int cR, int cC, int cB, int biasB,
                    int act, int batch) {
        k_gemm_wmma<<<dim3(M / 32, (N + 31) / 32, batch), 32, 0, stream>>>(
            A, B, C, bias, resid, K, N, aR, aC, aB, bR, bC, bB, bMod, cR, cC, cB, biasB,
            act);
    };

    auto resblock = [&](const float* src, float* dst, int pbase) {
        k_groupnorm_silu<<<256, 256, 0, stream>>>(src, in(pbase + 0), in(pbase + 1), W2);
        k_conv3x3_wmma<<<dim3(256, 8), 32, 0, stream>>>(W2, in(pbase + 2), in(pbase + 3),
                                                        nullptr, W3);
        k_groupnorm_silu<<<256, 256, 0, stream>>>(W3, in(pbase + 4), in(pbase + 5), W2);
        k_conv3x3_wmma<<<dim3(256, 8), 32, 0, stream>>>(W2, in(pbase + 6), in(pbase + 7),
                                                        src, dst);
    };

    auto vss = [&](float* xb, int p) {  // xb = W4 (B,L,256), updated in place
        k_layernorm<<<8192, 128, 0, stream>>>(xb, in(p + 0), in(p + 1), W5, 256, 256,
                                              nullptr, 0, 0);
        // xz = ln1 @ in_proj.T : [8192x256]x[256x1024] -> W6
        gemm(W5, in(p + 2), W6, nullptr, nullptr, 8192, 1024, 256,
             256, 1, 0, 1, 256, 0, 1, 1024, 1, 0, 0, 0, 1);
        // depthwise conv + silu -> W10 (B,L,512)
        k_dwconv_silu<<<16384, 256, 0, stream>>>(W6, in(p + 3), in(p + 4), W10);
        // 4 scan directions -> W7 (B,K,L,512)
        k_build_xs<<<16384, 256, 0, stream>>>(W10, W7);
        // x_dbl: batch 32 (b,k): [1024x512]x[512x48] -> W9 (B,K,48,L)
        gemm(W7, in(p + 5), W9, nullptr, nullptr, 1024, 48, 512,
             512, 1, 1 << 19, 1, 512, 48 * 512, 4, 1, 1024, 48 * 1024, 0, 0, 32);
        // delta = softplus(dts @ dt_w.T + dt_b): batch 32: [1024x16]x[16x512] -> W8
        gemm(W9, in(p + 6), W8, in(p + 7), nullptr, 1024, 512, 16,
             1, 1024, 48 * 1024, 1, 16, 512 * 16, 4, 512, 1, 1 << 19, 512, 2, 32);
        // selective scan (ys overwrites delta in W8)
        k_scan<<<64, 256, 0, stream>>>(W7, W8, W9, in(p + 8), in(p + 9));
        // merge directions -> W10 as (B,L,512)
        k_merge<<<16384, 256, 0, stream>>>(W8, W10);
        // out-norm LN * silu(z) -> W11 ; z = W6[..., 512:1024]
        k_layernorm<<<8192, 128, 0, stream>>>(W10, in(p + 10), in(p + 11), W11, 512, 512,
                                              W6, 1024, 512);
        // x += gated @ out_proj.T : [8192x512]x[512x256] -> xb (resid = xb)
        gemm(W11, in(p + 12), xb, nullptr, xb, 8192, 256, 512,
             512, 1, 0, 1, 512, 0, 1, 256, 1, 0, 0, 0, 1);
        k_layernorm<<<8192, 128, 0, stream>>>(xb, in(p + 13), in(p + 14), W5, 256, 256,
                                              nullptr, 0, 0);
        // fc1 + gelu -> W6 ; fc2 + bias + resid(xb) -> xb
        gemm(W5, in(p + 15), W6, in(p + 16), nullptr, 8192, 1024, 256,
             256, 1, 0, 1, 256, 0, 1, 1024, 1, 0, 0, 1, 1);
        gemm(W6, in(p + 17), xb, in(p + 18), xb, 8192, 256, 1024,
             1024, 1, 0, 1, 1024, 0, 1, 256, 1, 0, 0, 0, 1);
    };

    float* out = (float*)d_out;
    resblock(in(0), W0, 1);
    for (int l = 0; l < 2; ++l) {
        int vbase = 9 + l * 27;
        int rbase = 28 + l * 27;
        k_bchw_to_bhwc<<<8192, 256, 0, stream>>>(W0, W4);
        vss(W4, vbase);
        k_bhwc_to_bchw<<<8192, 256, 0, stream>>>(W4, W1);
        resblock(W1, (l == 1) ? out : W0, rbase);
    }
}
